// MultiScaleDecoder_86397562126380
// MI455X (gfx1250) — matrix operand order repaired and checked by reading, not compiled, not run on hardware
//
#include <hip/hip_runtime.h>
#include <hip/hip_bf16.h>
#include <math.h>

typedef __bf16 bf16_t;
typedef __attribute__((ext_vector_type(16))) __bf16 v16bf;
typedef __attribute__((ext_vector_type(8)))  __bf16 v8bf;
typedef __attribute__((ext_vector_type(4)))  __bf16 v4bf;
typedef __attribute__((ext_vector_type(8)))  float  v8f;

#define NPASS_MAX 4

struct GemmArgs {
  const float* A[NPASS_MAX];   // A plane per pass, row-major, k-stride 1
  const float* B[NPASS_MAX];   // B base per pass, strides bs_k/bs_n
  float sign[NPASS_MAX];       // applied to A while staging to LDS
  int   K[NPASS_MAX];          // reduction depth per pass
  int   asm_[NPASS_MAX];       // A row stride per pass
  int   npass;
  int   M, N;
  int   bs_k, bs_n;            // B element = B[k*bs_k + n*bs_n]
  long  azh, bzh, czh;         // per-blockIdx.z base offsets (elements)
  float* C;
  int   cs_m, cs_n;
  const float* bias;           // interleaved [N,2]: bias[2*n] (nullable)
  const float* bias2;          // second bias (nullable)
  float alpha;
};

// ---------------------------------------------------------------------------
// Generic multi-pass real GEMM with bf16 WMMA (f32 accumulate).
// Block: 256 threads = 8 waves. Block tile 64(M) x 64(N); wave grid 4x2,
// each wave computes a 16x32 strip = 2 WMMA tiles sharing one A operand.
// K chunked by 32 with zero padding. Fast (branch-free, vectorized) staging
// when the tile is fully in range; guarded fallback otherwise.
// ---------------------------------------------------------------------------
__launch_bounds__(256)
__global__ void gemm_bf16_wmma_kernel(GemmArgs g) {
  __shared__ bf16_t lA[64][40];   // [m][k], row stride 80B (16B aligned)
  __shared__ bf16_t lB[64][40];   // [n][k]

  const int tid   = threadIdx.x;
  const int wave  = tid >> 5;
  const int lane  = tid & 31;
  const int wm    = wave >> 1;        // 0..3 : M subtile (16 rows)
  const int wn    = wave & 1;         // 0..1 : N strip (32 cols)
  const int m0    = blockIdx.y * 64;
  const int n0    = blockIdx.x * 64;
  const int z     = blockIdx.z;
  const int lhalf = (lane >> 4) & 1;  // lane half selects K sub-blocks
  const int lmod  = lane & 15;

  v8f acc0 = {0.f, 0.f, 0.f, 0.f, 0.f, 0.f, 0.f, 0.f};
  v8f acc1 = {0.f, 0.f, 0.f, 0.f, 0.f, 0.f, 0.f, 0.f};

  for (int p = 0; p < g.npass; ++p) {
    const float* Ap  = g.A[p] + (long)z * g.azh;
    const float* Bp  = g.B[p] + (long)z * g.bzh;
    const float  sgn = g.sign[p];
    const int    Kp  = g.K[p];
    const int    am  = g.asm_[p];
    for (int k0 = 0; k0 < Kp; k0 += 32) {
      const bool fullA = (m0 + 64 <= g.M) && (k0 + 32 <= Kp);
      const bool fullB = (n0 + 64 <= g.N) && (k0 + 32 <= Kp);
      __syncthreads();
      // ---- stage A tile (64 x 32) -> bf16, sign folded in ----
      if (fullA) {
#pragma unroll
        for (int it = 0; it < 2; ++it) {
          const int idx = (tid + it * 256) * 4;       // 4 consecutive k
          const int r = idx >> 5, k = idx & 31;
          const float4 v = *(const float4*)(Ap + (long)(m0 + r) * am + (k0 + k));
          v4bf o = { (bf16_t)(sgn * v.x), (bf16_t)(sgn * v.y),
                     (bf16_t)(sgn * v.z), (bf16_t)(sgn * v.w) };
          *(v4bf*)(&lA[r][k]) = o;
        }
      } else {
        for (int idx = tid; idx < 64 * 32; idx += 256) {
          const int r = idx >> 5, k = idx & 31;
          const int gm = m0 + r, gk = k0 + k;
          float v = 0.0f;
          if (gm < g.M && gk < Kp) v = Ap[(long)gm * am + gk];
          lA[r][k] = (bf16_t)(sgn * v);
        }
      }
      // ---- stage B tile (64 n x 32 k), stored transposed [n][k] ----
      if (fullB) {
        if (g.bs_k == 1) {            // k-contiguous source (score GEMMs)
#pragma unroll
          for (int it = 0; it < 2; ++it) {
            const int idx = (tid + it * 256) * 4;
            const int n = idx >> 5, k = idx & 31;
            const float4 v = *(const float4*)(Bp + (long)(n0 + n) * g.bs_n + (k0 + k));
            v4bf o = { (bf16_t)v.x, (bf16_t)v.y, (bf16_t)v.z, (bf16_t)v.w };
            *(v4bf*)(&lB[n][k]) = o;
          }
        } else {
#pragma unroll
          for (int it = 0; it < 8; ++it) {
            const int idx = tid + it * 256;
            const int n = idx >> 5, k = idx & 31;
            lB[n][k] = (bf16_t)Bp[(long)(k0 + k) * g.bs_k + (long)(n0 + n) * g.bs_n];
          }
        }
      } else {
        for (int idx = tid; idx < 64 * 32; idx += 256) {
          const int n = idx >> 5, k = idx & 31;
          const int gn = n0 + n, gk = k0 + k;
          float v = 0.0f;
          if (gn < g.N && gk < Kp) v = Bp[(long)gk * g.bs_k + (long)gn * g.bs_n];
          lB[n][k] = (bf16_t)v;
        }
      }
      __syncthreads();
      // ---- per-wave WMMA: one A operand, two B operands, two wmma ----
      // A: lane m=lmod, K = lhalf*16 + 0..15
      const bf16_t* pa = &lA[wm * 16 + lmod][lhalf * 16];
      v8bf a0 = *(const v8bf*)(pa);
      v8bf a1 = *(const v8bf*)(pa + 8);
      v16bf av = __builtin_shufflevector(a0, a1, 0,1,2,3,4,5,6,7,8,9,10,11,12,13,14,15);
      // B: lane n=lmod (+16), K = lhalf*16 + 0..15 (contiguous in [n][k])
      const bf16_t* pb0 = &lB[wn * 32 + lmod][lhalf * 16];
      v8bf b00 = *(const v8bf*)(pb0);
      v8bf b01 = *(const v8bf*)(pb0 + 8);
      const bf16_t* pb1 = &lB[wn * 32 + 16 + lmod][lhalf * 16];
      v8bf b10 = *(const v8bf*)(pb1);
      v8bf b11 = *(const v8bf*)(pb1 + 8);
      v16bf bv0 = __builtin_shufflevector(b00, b01, 0,1,2,3,4,5,6,7,8,9,10,11,12,13,14,15);
      v16bf bv1 = __builtin_shufflevector(b10, b11, 0,1,2,3,4,5,6,7,8,9,10,11,12,13,14,15);
      acc0 = __builtin_amdgcn_wmma_f32_16x16x32_bf16(
          false, av, false, bv0, (short)0, acc0, false, false);
      acc1 = __builtin_amdgcn_wmma_f32_16x16x32_bf16(
          false, av, false, bv1, (short)0, acc1, false, false);
    }
  }

  // Epilogue: C/D layout — VGPR r: M = r + lhalf*8, N = lmod
  float* Cp = g.C + (long)z * g.czh;
#pragma unroll
  for (int half = 0; half < 2; ++half) {
    const int ncol = n0 + wn * 32 + half * 16 + lmod;
    const v8f& acc = half ? acc1 : acc0;
    if (ncol < g.N) {
      float bsum = 0.0f;
      if (g.bias)  bsum += g.bias[2 * ncol];
      if (g.bias2) bsum += g.bias2[2 * ncol];
#pragma unroll
      for (int r = 0; r < 8; ++r) {
        const int mrow = m0 + wm * 16 + r + lhalf * 8;
        if (mrow < g.M)
          Cp[(long)mrow * g.cs_m + (long)ncol * g.cs_n] = g.alpha * acc[r] + bsum;
      }
    }
  }
}

// ---------------------------------------------------------------------------
// Bilinear resize along L (half-pixel centers), planar complex.
// ---------------------------------------------------------------------------
__global__ void resize_bilinear_kernel(const float* __restrict__ inr,
                                       const float* __restrict__ ini,
                                       float* __restrict__ outr,
                                       float* __restrict__ outi,
                                       int Bn, int Lin, int Lout, int D) {
  long idx = (long)blockIdx.x * blockDim.x + threadIdx.x;
  long total = (long)Bn * Lout * D;
  if (idx >= total) return;
  int d = (int)(idx % D);
  long t = idx / D;
  int l = (int)(t % Lout);
  int b = (int)(t / Lout);
  float c = (l + 0.5f) * (float)Lin / (float)Lout - 0.5f;
  float fl = floorf(c);
  int l0 = (int)fl;
  float w = c - fl;
  int l0c = min(max(l0, 0), Lin - 1);
  int l1c = min(max(l0 + 1, 0), Lin - 1);
  long base = (long)b * Lin * D + d;
  outr[idx] = inr[base + (long)l0c * D] * (1.0f - w) + inr[base + (long)l1c * D] * w;
  outi[idx] = ini[base + (long)l0c * D] * (1.0f - w) + ini[base + (long)l1c * D] * w;
}

// ---------------------------------------------------------------------------
// Complex LayerNorm (+ optional fused modReLU). One block per token row,
// blockDim == D (64 or 128). In-place safe.
// ---------------------------------------------------------------------------
__global__ void cln_modrelu_kernel(const float* __restrict__ xr,
                                   const float* __restrict__ xi,
                                   const float* __restrict__ gg,   // [D,2]
                                   const float* __restrict__ bb,   // [D,2]
                                   const float* __restrict__ mb,   // [D] or null
                                   float* __restrict__ outr,
                                   float* __restrict__ outi,
                                   int D, int ostride, int do_modrelu) {
  __shared__ float red[256];
  const int d = threadIdx.x;
  const long base = (long)blockIdx.x * D + d;
  float vr = xr[base];
  float vi = xi[base];

  auto bsum = [&](float v) -> float {
    red[d] = v;
    __syncthreads();
    for (int s = blockDim.x >> 1; s > 0; s >>= 1) {
      if (d < s) red[d] += red[d + s];
      __syncthreads();
    }
    float r = red[0];
    __syncthreads();
    return r;
  };

  const float invD = 1.0f / (float)D;
  float mr = bsum(vr) * invD;
  float mi = bsum(vi) * invD;
  float cr = vr - mr, ci = vi - mi;
  float var = bsum(cr * cr + ci * ci) * invD;
  float inv = rsqrtf(var + 1e-6f);
  float gr = gg[2 * d], gi = gg[2 * d + 1];
  float br = bb[2 * d], bi = bb[2 * d + 1];
  float nr = cr * inv, ni = ci * inv;
  float yr = nr * gr - ni * gi + br;
  float yi = nr * gi + ni * gr + bi;
  if (do_modrelu) {
    float mag = sqrtf(yr * yr + yi * yi);
    float f = fmaxf(mag + mb[d], 0.0f) / (mag + 1e-6f);
    yr *= f;
    yi *= f;
  }
  outr[base * ostride] = yr;
  outi[base * ostride] = yi;
}

// ---------------------------------------------------------------------------
// Row softmax over length L (one block per row).
// ---------------------------------------------------------------------------
__global__ void softmax_rows_kernel(float* __restrict__ S, int L) {
  __shared__ float red[256];
  const int tid = threadIdx.x;
  float* row = S + (long)blockIdx.x * L;
  float m = -3.4e38f;
  for (int j = tid; j < L; j += 256) m = fmaxf(m, row[j]);
  red[tid] = m;
  __syncthreads();
  for (int s = 128; s > 0; s >>= 1) {
    if (tid < s) red[tid] = fmaxf(red[tid], red[tid + s]);
    __syncthreads();
  }
  m = red[0];
  __syncthreads();
  float acc = 0.0f;
  for (int j = tid; j < L; j += 256) {
    float e = __expf(row[j] - m);
    row[j] = e;
    acc += e;
  }
  red[tid] = acc;
  __syncthreads();
  for (int s = 128; s > 0; s >>= 1) {
    if (tid < s) red[tid] += red[tid + s];
    __syncthreads();
  }
  float inv = 1.0f / red[0];
  __syncthreads();
  for (int j = tid; j < L; j += 256) row[j] *= inv;
}

// ---------------------------------------------------------------------------
// Host orchestration
// ---------------------------------------------------------------------------
extern "C" void kernel_launch(void* const* d_in, const int* in_sizes, int n_in,
                              void* d_out, int out_size, void* d_ws, size_t ws_size,
                              hipStream_t stream) {
  (void)in_sizes; (void)n_in; (void)out_size; (void)ws_size;
  const int Bn = 8, H = 8;

  const float* emb0_r = (const float*)d_in[0];
  const float* emb0_i = (const float*)d_in[1];
  const float* emb1_r = (const float*)d_in[2];
  const float* emb1_i = (const float*)d_in[3];
  const float* emb2_r = (const float*)d_in[4];
  const float* emb2_i = (const float*)d_in[5];
  // d_in[6] = t (unused by the reference body)

  // params flattened in dict insertion order
  enum { PWh = 0, Pbh, PWs, Pbs, Pga, Pba, Pmb, PWq, Pbq, PWk, Pbk,
         PWv, Pbv, PWo, Pbo, Pgb, Pbbx };
  const int P1b = 7, P2b = 24;
  auto P = [&](int base, int o) { return (const float*)d_in[base + o]; };

  // workspace arena (floats): 16,777,216 floats = 64 MB total, aliased across blocks
  float* ws = (float*)d_ws;
  size_t off = 0;
  auto alloc = [&](size_t n) { float* p = ws + off; off += n; return p; };
  float* RZr = alloc(1048576);  float* RZi = alloc(1048576);   // resized input
  float* Xr  = alloc(524288);   float* Xi  = alloc(524288);    // post-linear / post-LN
  float* Qr  = alloc(524288);   float* Qi  = alloc(524288);
  float* Kr_ = alloc(524288);   float* Ki  = alloc(524288);
  float* Vr  = alloc(524288);   float* Vi  = alloc(524288);
  float* Or_ = alloc(524288);   float* Oi_ = alloc(524288);    // attention out
  float* Yr  = alloc(524288);   float* Yi  = alloc(524288);    // block output
  float* Sb  = alloc(8388608);                                  // scores (per-b reuse)

  auto launch_gemm = [&](const GemmArgs& a, int gz) {
    dim3 grid((a.N + 63) / 64, (a.M + 63) / 64, gz);
    gemm_bf16_wmma_kernel<<<grid, dim3(256), 0, stream>>>(a);
  };

  // complex GEMM: Y = X1@W1 [+ X2@W2] + b1 [+ b2]; W interleaved [K,N,2]
  auto cgemm = [&](const float* X1r, const float* X1i, const float* W1, int K1,
                   const float* X2r, const float* X2i, const float* W2, int K2,
                   const float* b1, const float* b2,
                   int M, int N, float* Cr, float* Ci) {
    GemmArgs a = {};
    a.M = M; a.N = N;
    a.bs_k = 2 * N; a.bs_n = 2;
    a.cs_m = N; a.cs_n = 1;
    a.azh = a.bzh = a.czh = 0;
    a.alpha = 1.0f;
    a.npass = X2r ? 4 : 2;
    // real: Xr@Wr - Xi@Wi
    a.A[0] = X1r; a.B[0] = W1 + 0; a.sign[0] =  1.0f; a.K[0] = K1; a.asm_[0] = K1;
    a.A[1] = X1i; a.B[1] = W1 + 1; a.sign[1] = -1.0f; a.K[1] = K1; a.asm_[1] = K1;
    if (X2r) {
      a.A[2] = X2r; a.B[2] = W2 + 0; a.sign[2] =  1.0f; a.K[2] = K2; a.asm_[2] = K2;
      a.A[3] = X2i; a.B[3] = W2 + 1; a.sign[3] = -1.0f; a.K[3] = K2; a.asm_[3] = K2;
    }
    a.C = Cr;
    a.bias  = b1;
    a.bias2 = b2;
    launch_gemm(a, 1);
    // imag: Xr@Wi + Xi@Wr
    a.B[0] = W1 + 1; a.sign[0] = 1.0f;
    a.B[1] = W1 + 0; a.sign[1] = 1.0f;
    if (X2r) { a.B[2] = W2 + 1; a.sign[2] = 1.0f; a.B[3] = W2 + 0; a.sign[3] = 1.0f; }
    a.C = Ci;
    a.bias  = b1 ? b1 + 1 : nullptr;
    a.bias2 = b2 ? b2 + 1 : nullptr;
    launch_gemm(a, 1);
  };

  auto attention = [&](int L, int D, int hd,
                       const float* hr, const float* hi,
                       const float* Wq, const float* bq,
                       const float* Wk, const float* bk,
                       const float* Wv, const float* bv,
                       const float* Wo, const float* bo,
                       float* outR, float* outI) {
    const int M = Bn * L;
    cgemm(hr, hi, Wq, D, nullptr, nullptr, nullptr, 0, bq, nullptr, M, D, Qr, Qi);
    cgemm(hr, hi, Wk, D, nullptr, nullptr, nullptr, 0, bk, nullptr, M, D, Kr_, Ki);
    cgemm(hr, hi, Wv, D, nullptr, nullptr, nullptr, 0, bv, nullptr, M, D, Vr, Vi);
    const float scale = 1.0f / sqrtf((float)hd);
    for (int b = 0; b < Bn; ++b) {
      const long pb = (long)b * L * D;
      // scores[h,l,m] = scale * (Qr.Kr^T + Qi.Ki^T); z = head
      GemmArgs s = {};
      s.npass = 2; s.M = L; s.N = L;
      s.A[0] = Qr + pb; s.B[0] = Kr_ + pb; s.sign[0] = 1.0f; s.K[0] = hd; s.asm_[0] = D;
      s.A[1] = Qi + pb; s.B[1] = Ki  + pb; s.sign[1] = 1.0f; s.K[1] = hd; s.asm_[1] = D;
      s.bs_k = 1; s.bs_n = D;           // B[k=d][n=m] = K[m, h*hd+d]  (free transpose)
      s.azh = hd; s.bzh = hd; s.czh = (long)L * L;
      s.C = Sb; s.cs_m = L; s.cs_n = 1;
      s.alpha = scale;
      launch_gemm(s, H);
      softmax_rows_kernel<<<dim3(H * L), dim3(256), 0, stream>>>(Sb, L);
      // O = P @ V (real and imag separately); z = head
      GemmArgs ap = {};
      ap.npass = 1; ap.M = L; ap.N = hd;
      ap.A[0] = Sb; ap.sign[0] = 1.0f; ap.K[0] = L; ap.asm_[0] = L;
      ap.bs_k = D; ap.bs_n = 1;
      ap.azh = (long)L * L; ap.bzh = hd; ap.czh = hd;
      ap.cs_m = D; ap.cs_n = 1; ap.alpha = 1.0f;
      ap.B[0] = Vr + pb; ap.C = Or_ + pb; launch_gemm(ap, H);
      ap.B[0] = Vi + pb; ap.C = Oi_ + pb; launch_gemm(ap, H);
    }
    cgemm(Or_, Oi_, Wo, D, nullptr, nullptr, nullptr, 0, bo, nullptr, M, D, outR, outI);
  };

  // -------- Block 1: [8,256,256] -> [8,512,128] --------
  {
    long total = (long)Bn * 512 * 256;
    resize_bilinear_kernel<<<dim3((unsigned)((total + 255) / 256)), dim3(256), 0, stream>>>(
        emb2_r, emb2_i, RZr, RZi, Bn, 256, 512, 256);
    cgemm(RZr, RZi, P(P1b, PWh), 256, emb1_r, emb1_i, P(P1b, PWs), 128,
          P(P1b, Pbh), P(P1b, Pbs), 4096, 128, Xr, Xi);
    cln_modrelu_kernel<<<dim3(4096), dim3(128), 0, stream>>>(
        Xr, Xi, P(P1b, Pga), P(P1b, Pba), P(P1b, Pmb), Xr, Xi, 128, 1, 1);
    attention(512, 128, 16, Xr, Xi,
              P(P1b, PWq), P(P1b, Pbq), P(P1b, PWk), P(P1b, Pbk),
              P(P1b, PWv), P(P1b, Pbv), P(P1b, PWo), P(P1b, Pbo), Yr, Yi);
    cln_modrelu_kernel<<<dim3(4096), dim3(128), 0, stream>>>(
        Yr, Yi, P(P1b, Pgb), P(P1b, Pbbx), nullptr, Yr, Yi, 128, 1, 0);
  }

  // -------- Block 2: [8,512,128] -> [8,1024,64] --------
  {
    long total = (long)Bn * 1024 * 128;
    resize_bilinear_kernel<<<dim3((unsigned)((total + 255) / 256)), dim3(256), 0, stream>>>(
        Yr, Yi, RZr, RZi, Bn, 512, 1024, 128);
    cgemm(RZr, RZi, P(P2b, PWh), 128, emb0_r, emb0_i, P(P2b, PWs), 64,
          P(P2b, Pbh), P(P2b, Pbs), 8192, 64, Xr, Xi);
    cln_modrelu_kernel<<<dim3(8192), dim3(64), 0, stream>>>(
        Xr, Xi, P(P2b, Pga), P(P2b, Pba), P(P2b, Pmb), Xr, Xi, 64, 1, 1);
    attention(1024, 64, 8, Xr, Xi,
              P(P2b, PWq), P(P2b, Pbq), P(P2b, PWk), P(P2b, Pbk),
              P(P2b, PWv), P(P2b, Pbv), P(P2b, PWo), P(P2b, Pbo), Yr, Yi);
    float* out = (float*)d_out;  // complex64 layout: interleaved (r,i)
    cln_modrelu_kernel<<<dim3(8192), dim3(64), 0, stream>>>(
        Yr, Yi, P(P2b, Pgb), P(P2b, Pbbx), nullptr, out, out + 1, 64, 2, 0);
  }
}
